// MultiHeadDotProductAttention_14980845928970
// MI455X (gfx1250) — matrix operand order repaired and checked by reading, not compiled, not run on hardware
//
#include <hip/hip_runtime.h>

// ---------------------------------------------------------------------------
// MI455X (gfx1250) multi-head (multi-query) attention with RoPE.
// All GEMMs via v_wmma_f32_16x16x32_bf16 (wave32, 16x16 tiles, K=32).
// Attention stages K/V chunks into LDS via async global->LDS DMA,
// double-buffered, shared by the 4 waves of a block.
// ---------------------------------------------------------------------------

typedef __attribute__((ext_vector_type(16))) __bf16 v16bf;
typedef __attribute__((ext_vector_type(8)))  __bf16 v8bf;
typedef __attribute__((ext_vector_type(4)))  __bf16 v4bf;
typedef __attribute__((ext_vector_type(8)))  float  v8f;
typedef __attribute__((ext_vector_type(4)))  int    v4i;

#define B_   2
#define L_   2048
#define F_   1024
#define H_   16
#define D_   64
#define HD_  1024
#define BIGNEG (-3.0e38f)

#if __has_builtin(__builtin_amdgcn_global_load_async_to_lds_b128)
#define HAVE_ASYNC_LDS 1
#else
#define HAVE_ASYNC_LDS 0
#endif

__device__ __forceinline__ v8f wmma_bf16(v16bf a, v16bf b, v8f c) {
  return __builtin_amdgcn_wmma_f32_16x16x32_bf16(false, a, false, b, (short)0, c,
                                                 false, false);
}

// Copy 16B (8 bf16) global -> LDS. Async (ASYNCcnt-tracked) when available.
__device__ __forceinline__ void copy16_g2l(const __bf16* g, __bf16* l) {
#if HAVE_ASYNC_LDS
  __builtin_amdgcn_global_load_async_to_lds_b128(
      (__attribute__((address_space(1))) v4i*)(v4i*)(void*)g,
      (__attribute__((address_space(3))) v4i*)(v4i*)(void*)l, 0, 0);
#else
  *(v8bf*)l = *(const v8bf*)g;
#endif
}

__device__ __forceinline__ void wait_async0() {
#if HAVE_ASYNC_LDS
#if __has_builtin(__builtin_amdgcn_s_wait_asynccnt)
  __builtin_amdgcn_s_wait_asynccnt(0);
#else
  asm volatile("s_wait_asynccnt 0" ::: "memory");
#endif
#endif
}

// A operand (16x32, 16-bit): lane holds row m = lane&15; lanes 0-15 carry
// K = [k0, k0+7] & [k0+16, k0+23]; lanes 16-31 carry K = [k0+8..15] & [k0+24..31].
__device__ __forceinline__ v16bf load_A16x32(const __bf16* A, int ld, int m, int hf, int k0) {
  const __bf16* p = A + (size_t)m * ld + k0 + 8 * hf;
  v8bf lo = *(const v8bf*)p;
  v8bf hi = *(const v8bf*)(p + 16);
  v16bf r;
#pragma unroll
  for (int j = 0; j < 8; ++j) { r[j] = lo[j]; r[j + 8] = hi[j]; }
  return r;
}

// B operand (32x16, 16-bit) from a K-contiguous ("BT") layout: BT[n][k].
// lane holds column n = lane&15; lanes 0-15: K=[k0..k0+7] & [k0+16..k0+23], lanes 16-31: K=[k0+8..15] & [k0+24..31].
__device__ __forceinline__ v16bf load_B32x16(const __bf16* BT, int ld, int n, int hf, int k0) {
  const __bf16* p = BT + (size_t)n * ld + k0 + 8 * hf;
  v8bf lo = *(const v8bf*)p;
  v8bf hi = *(const v8bf*)(p + 16);
  v16bf r;
#pragma unroll
  for (int j = 0; j < 8; ++j) { r[j] = lo[j]; r[j + 8] = hi[j]; }
  return r;
}

// C/D layout: lane L -> col n = L&15, VGPR i -> row i + 8*(L>>4).

__device__ __forceinline__ float rowmax16(float v) {
  v = fmaxf(v, __shfl_xor(v, 1, 32));
  v = fmaxf(v, __shfl_xor(v, 2, 32));
  v = fmaxf(v, __shfl_xor(v, 4, 32));
  v = fmaxf(v, __shfl_xor(v, 8, 32));
  return v;
}
__device__ __forceinline__ float rowsum16(float v) {
  v += __shfl_xor(v, 1, 32);
  v += __shfl_xor(v, 2, 32);
  v += __shfl_xor(v, 4, 32);
  v += __shfl_xor(v, 8, 32);
  return v;
}

// RoPE on a C-layout tile. d = feature index within rotary span (pairs (2j,2j+1)),
// rows of the tile are consecutive sequence positions starting at pos0.
__device__ __forceinline__ void rope_tile(v8f& c, int d, int pos0, int hf,
                                          const float* sinT, const float* cosT) {
  const int j = d >> 1;
  const bool odd = (d & 1) != 0;
#pragma unroll
  for (int i = 0; i < 8; ++i) {
    float x = c[i];
    float partner = __shfl_xor(x, 1, 32);   // value at d^1 (adjacent lane)
    int pos = pos0 + i + 8 * hf;
    float s  = sinT[pos * 32 + j];
    float co = cosT[pos * 32 + j];
    c[i] = odd ? (x * co + partner * s) : (x * co - partner * s);
  }
}

// ------------------------------- prep kernels ------------------------------

__global__ void mha_cvt_bf16(const float* __restrict__ src, __bf16* __restrict__ dst, int n) {
  int i = (blockIdx.x * blockDim.x + threadIdx.x) * 4;
  if (i >= n) return;
  float4 v = *(const float4*)(src + i);
  v4bf o;
  o[0] = (__bf16)v.x; o[1] = (__bf16)v.y; o[2] = (__bf16)v.z; o[3] = (__bf16)v.w;
  *(v4bf*)(dst + i) = o;
}

// dst[c][r] = (bf16)src[r][c]; R, C multiples of 32.
__global__ void mha_transpose_cvt(const float* __restrict__ src, __bf16* __restrict__ dst,
                                  int R, int C) {
  __shared__ float tile[32][33];
  int bx = blockIdx.x * 32;          // col base in src
  int by = blockIdx.y * 32;          // row base in src
  int tx = threadIdx.x & 31, ty = threadIdx.x >> 5;   // 256 threads
#pragma unroll
  for (int k = 0; k < 32; k += 8)
    tile[ty + k][tx] = src[(size_t)(by + ty + k) * C + bx + tx];
  __syncthreads();
#pragma unroll
  for (int k = 0; k < 32; k += 8)
    dst[(size_t)(bx + ty + k) * R + by + tx] = (__bf16)tile[tx][ty + k];
}

__global__ void mha_rope_table(float* __restrict__ sinT, float* __restrict__ cosT, int L) {
  int idx = blockIdx.x * blockDim.x + threadIdx.x;
  if (idx >= L * 32) return;
  int pos = idx >> 5, j = idx & 31;
  float inv_freq = powf(10000.0f, -(2.0f * (float)j) / 64.0f);
  float ang = (float)pos * inv_freq;
  sinT[idx] = sinf(ang);
  cosT[idx] = cosf(ang);
}

// --------------------------- K/V projection + RoPE(K) ----------------------
// One wave -> 16 kv rows. K stored [b][l][d] (row-major: B-operand layout for QK^T).
// V stored transposed [b][d][l] (B-operand layout for P*V).
__global__ __launch_bounds__(128) void mha_kv_proj(
    const __bf16* __restrict__ Xkv, const __bf16* __restrict__ WkT,
    const __bf16* __restrict__ WvT, const float* __restrict__ bk,
    const float* __restrict__ bv, const float* __restrict__ sinT,
    const float* __restrict__ cosT, __bf16* __restrict__ Kd, __bf16* __restrict__ Vt) {
  int w    = (blockIdx.x * blockDim.x + threadIdx.x) >> 5;   // 0..255 row-tiles
  int lane = threadIdx.x & 31;
  int m = lane & 15, n = m, hf = lane >> 4;
  int r0 = w * 16;
  int bidx = r0 / L_;
  int l0 = r0 % L_;

  v8f ck[4] = {};
  v8f cv[4] = {};
  const __bf16* Arow = Xkv + (size_t)r0 * F_;
  for (int k0 = 0; k0 < F_; k0 += 32) {
    v16bf a = load_A16x32(Arow, F_, m, hf, k0);
#pragma unroll
    for (int t = 0; t < 4; ++t) {
      v16bf bkm = load_B32x16(WkT + (size_t)(t * 16) * F_, F_, n, hf, k0);
      ck[t] = wmma_bf16(a, bkm, ck[t]);
      v16bf bvm = load_B32x16(WvT + (size_t)(t * 16) * F_, F_, n, hf, k0);
      cv[t] = wmma_bf16(a, bvm, cv[t]);
    }
  }
#pragma unroll
  for (int t = 0; t < 4; ++t) {
    int d = t * 16 + n;
    float bkd = bk[d], bvd = bv[d];
#pragma unroll
    for (int i = 0; i < 8; ++i) { ck[t][i] += bkd; cv[t][i] += bvd; }
    rope_tile(ck[t], d, l0, hf, sinT, cosT);
#pragma unroll
    for (int i = 0; i < 8; ++i)
      Kd[((size_t)bidx * L_ + l0 + i + 8 * hf) * D_ + d] = (__bf16)ck[t][i];
    v8bf vv;
#pragma unroll
    for (int i = 0; i < 8; ++i) vv[i] = (__bf16)cv[t][i];
    *(v8bf*)(Vt + ((size_t)bidx * D_ + d) * L_ + l0 + 8 * hf) = vv;
  }
}

// --------------------------- Q projection + RoPE + scale -------------------
// One wave -> 16 q rows x 1 head. Q stored [b][h][l][d] (A-operand layout).
__global__ __launch_bounds__(128) void mha_q_proj(
    const __bf16* __restrict__ Xq, const __bf16* __restrict__ WqT,
    const float* __restrict__ bq, const float* __restrict__ sinT,
    const float* __restrict__ cosT, __bf16* __restrict__ Qd) {
  int w    = (blockIdx.x * blockDim.x + threadIdx.x) >> 5;   // 0..4095
  int lane = threadIdx.x & 31;
  int m = lane & 15, n = m, hf = lane >> 4;
  int hh = w & 15;
  int rt = w >> 4;                    // 0..255
  int r0 = rt * 16;
  int bidx = r0 / L_;
  int l0 = r0 % L_;

  v8f cq[4] = {};
  const __bf16* Arow = Xq + (size_t)r0 * F_;
  for (int k0 = 0; k0 < F_; k0 += 32) {
    v16bf a = load_A16x32(Arow, F_, m, hf, k0);
#pragma unroll
    for (int t = 0; t < 4; ++t) {
      v16bf bm = load_B32x16(WqT + (size_t)(hh * 64 + t * 16) * F_, F_, n, hf, k0);
      cq[t] = wmma_bf16(a, bm, cq[t]);
    }
  }
#pragma unroll
  for (int t = 0; t < 4; ++t) {
    int d = t * 16 + n;                     // within head
    float bqd = bq[hh * 64 + d];
#pragma unroll
    for (int i = 0; i < 8; ++i) cq[t][i] += bqd;
    rope_tile(cq[t], d, l0, hf, sinT, cosT);
#pragma unroll
    for (int i = 0; i < 8; ++i) {
      float v = cq[t][i] * 0.125f;          // 1/sqrt(64)
      Qd[(((size_t)bidx * H_ + hh) * L_ + l0 + i + 8 * hf) * D_ + d] = (__bf16)v;
    }
  }
}

// ------------------------------ flash attention ----------------------------
// One wave -> one (b, h, 16-row q tile). The 4 waves of a block share the same
// (b, h); K/V chunks (32 kv positions) are staged into LDS once per block via
// async global->LDS, double-buffered so DMA of chunk i+1 overlaps compute of i.
__global__ __launch_bounds__(128) void mha_attn(
    const __bf16* __restrict__ Qd, const __bf16* __restrict__ Kd,
    const __bf16* __restrict__ Vt, const unsigned char* __restrict__ msk,
    __bf16* __restrict__ O) {
  __shared__ __align__(16) __bf16 Kbuf[2][32][64];   // [buf][l][d]
  __shared__ __align__(16) __bf16 Vbuf[2][64][32];   // [buf][d][l]
  __shared__ __align__(16) __bf16 pshm[4][16][40];   // P staging, C->A transpose
  int tid   = threadIdx.x;
  int wslot = tid >> 5;
  int w     = (blockIdx.x * blockDim.x + tid) >> 5;  // 0..4095
  int lane  = tid & 31;
  int m = lane & 15, n = m, hf = lane >> 4;
  int qt   = w & 127;          // Lq/16 (4 consecutive per block -> same b,h)
  int hh   = (w >> 7) & 15;
  int bidx = w >> 11;

  const __bf16* Qb = Qd + (((size_t)bidx * H_ + hh) * L_ + qt * 16) * D_;
  v16bf aq0 = load_A16x32(Qb, D_, m, hf, 0);
  v16bf aq1 = load_A16x32(Qb, D_, m, hf, 32);

  const __bf16* Kb = Kd + (size_t)bidx * L_ * D_;
  const __bf16* Vb = Vt + (size_t)bidx * D_ * L_;
  const unsigned char* mb = msk + ((size_t)bidx * L_ + qt * 16) * L_;

  // Cooperative stage of one 32-kv chunk (K: 4KB contiguous, V: 64 x 64B rows).
  auto stage = [&](int buf, int kv0) {
    const __bf16* gk = Kb + (size_t)kv0 * D_ + tid * 16;
    __bf16* lk = &Kbuf[buf][0][0] + tid * 16;
    copy16_g2l(gk, lk);
    copy16_g2l(gk + 8, lk + 8);
    int dv = tid >> 1, part = tid & 1;
    const __bf16* gv = Vb + (size_t)dv * L_ + kv0 + part * 16;
    __bf16* lv = &Vbuf[buf][dv][part * 16];
    copy16_g2l(gv, lv);
    copy16_g2l(gv + 8, lv + 8);
  };

  float mrow[8], lrow[8];
#pragma unroll
  for (int i = 0; i < 8; ++i) { mrow[i] = BIGNEG; lrow[i] = 0.0f; }
  v8f o0 = {}, o1 = {}, o2 = {}, o3 = {};

  stage(0, 0);
  int cur = 0;
  for (int kv0 = 0; kv0 < L_; kv0 += 32) {
    wait_async0();       // this wave's async LDS writes complete
    __syncthreads();     // whole block's chunk visible; prev compute finished
    if (kv0 + 32 < L_) {
      stage(cur ^ 1, kv0 + 32);
      __builtin_prefetch(mb + (size_t)(lane & 15) * L_ + kv0 + 32, 0, 1);
    }
    // S = Q K^T  (two 16x16 tiles from LDS, K-dim = 64 -> 2 WMMAs each)
    v8f s0 = {}, s1 = {};
    s0 = wmma_bf16(aq0, load_B32x16(&Kbuf[cur][0][0], 64, n, hf, 0), s0);
    s0 = wmma_bf16(aq1, load_B32x16(&Kbuf[cur][0][0], 64, n, hf, 32), s0);
    s1 = wmma_bf16(aq0, load_B32x16(&Kbuf[cur][16][0], 64, n, hf, 0), s1);
    s1 = wmma_bf16(aq1, load_B32x16(&Kbuf[cur][16][0], 64, n, hf, 32), s1);
    // mask
#pragma unroll
    for (int i = 0; i < 8; ++i) {
      const unsigned char* mr = mb + (size_t)(i + 8 * hf) * L_ + kv0;
      if (!mr[n])      s0[i] = BIGNEG;
      if (!mr[16 + n]) s1[i] = BIGNEG;
    }
    // online softmax (rows live in 16-lane groups)
    float alpha[8];
#pragma unroll
    for (int i = 0; i < 8; ++i) {
      float rm = rowmax16(fmaxf(s0[i], s1[i]));
      float mn = fmaxf(mrow[i], rm);
      float a  = __expf(mrow[i] - mn);
      float p0 = __expf(s0[i] - mn);
      float p1 = __expf(s1[i] - mn);
      lrow[i] = lrow[i] * a + rowsum16(p0 + p1);
      mrow[i] = mn;
      alpha[i] = a;
      s0[i] = p0; s1[i] = p1;
    }
#pragma unroll
    for (int i = 0; i < 8; ++i) {
      o0[i] *= alpha[i]; o1[i] *= alpha[i]; o2[i] *= alpha[i]; o3[i] *= alpha[i];
    }
    // P: C layout -> A layout via LDS (per-wave region; LDS in-order per wave)
#pragma unroll
    for (int i = 0; i < 8; ++i) {
      pshm[wslot][i + 8 * hf][n]      = (__bf16)s0[i];
      pshm[wslot][i + 8 * hf][16 + n] = (__bf16)s1[i];
    }
    v16bf pa;
    {
      v8bf lo = *(const v8bf*)&pshm[wslot][m][8 * hf];
      v8bf hi = *(const v8bf*)&pshm[wslot][m][16 + 8 * hf];
#pragma unroll
      for (int j = 0; j < 8; ++j) { pa[j] = lo[j]; pa[j + 8] = hi[j]; }
    }
    // O += P V  (K-dim = 32, four 16-wide d tiles from LDS)
    o0 = wmma_bf16(pa, load_B32x16(&Vbuf[cur][0][0], 32, n, hf, 0), o0);
    o1 = wmma_bf16(pa, load_B32x16(&Vbuf[cur][16][0], 32, n, hf, 0), o1);
    o2 = wmma_bf16(pa, load_B32x16(&Vbuf[cur][32][0], 32, n, hf, 0), o2);
    o3 = wmma_bf16(pa, load_B32x16(&Vbuf[cur][48][0], 32, n, hf, 0), o3);
    __syncthreads();     // all waves done reading buf[cur] before reuse
    cur ^= 1;
  }
  // finalize + store O[b][l][h*64+d] (bf16, row-major for final projection)
#pragma unroll
  for (int i = 0; i < 8; ++i) {
    float inv = 1.0f / lrow[i];
    __bf16* op = O + ((size_t)bidx * L_ + qt * 16 + i + 8 * hf) * HD_ + hh * 64;
    op[0 + n]  = (__bf16)(o0[i] * inv);
    op[16 + n] = (__bf16)(o1[i] * inv);
    op[32 + n] = (__bf16)(o2[i] * inv);
    op[48 + n] = (__bf16)(o3[i] * inv);
  }
}

// ------------------------------ output projection --------------------------
// One wave -> 16 rows x 64 cols of out = O @ Wo + bo   (fp32 result).
__global__ __launch_bounds__(128) void mha_out_proj(
    const __bf16* __restrict__ O, const __bf16* __restrict__ WoT,
    const float* __restrict__ bo, float* __restrict__ out) {
  int w    = (blockIdx.x * blockDim.x + threadIdx.x) >> 5;   // 0..4095
  int lane = threadIdx.x & 31;
  int m = lane & 15, n = m, hf = lane >> 4;
  int fg = w & 15;
  int rt = w >> 4;
  int r0 = rt * 16;

  v8f c[4] = {};
  const __bf16* Arow = O + (size_t)r0 * HD_;
  for (int k0 = 0; k0 < HD_; k0 += 32) {
    v16bf a = load_A16x32(Arow, HD_, m, hf, k0);
#pragma unroll
    for (int t = 0; t < 4; ++t) {
      v16bf bm = load_B32x16(WoT + (size_t)(fg * 64 + t * 16) * HD_, HD_, n, hf, k0);
      c[t] = wmma_bf16(a, bm, c[t]);
    }
  }
#pragma unroll
  for (int t = 0; t < 4; ++t) {
    int f = fg * 64 + t * 16 + n;
    float bias = bo[f];
#pragma unroll
    for (int i = 0; i < 8; ++i)
      out[(size_t)(r0 + i + 8 * hf) * F_ + f] = c[t][i] + bias;
  }
}

// ------------------------------- launcher ----------------------------------

extern "C" void kernel_launch(void* const* d_in, const int* in_sizes, int n_in,
                              void* d_out, int out_size, void* d_ws, size_t ws_size,
                              hipStream_t stream) {
  (void)in_sizes; (void)n_in; (void)out_size; (void)ws_size;
  const float* inq  = (const float*)d_in[0];
  const float* inkv = (const float*)d_in[1];
  const unsigned char* msk = (const unsigned char*)d_in[2];  // jax bool -> 1 byte
  const float* Wq = (const float*)d_in[3];
  const float* bq = (const float*)d_in[4];
  const float* Wk = (const float*)d_in[5];
  const float* bk = (const float*)d_in[6];
  const float* Wv = (const float*)d_in[7];
  const float* bv = (const float*)d_in[8];
  const float* Wo = (const float*)d_in[9];
  const float* bo = (const float*)d_in[10];
  float* out = (float*)d_out;

  char* wsp = (char*)d_ws;
  auto take = [&](size_t bytes) -> char* {
    char* p = wsp; wsp += (bytes + 255) & ~(size_t)255; return p;
  };
  __bf16* Xq   = (__bf16*)take((size_t)B_ * L_ * F_ * 2);
  __bf16* Xkv  = (__bf16*)take((size_t)B_ * L_ * F_ * 2);
  __bf16* WqT  = (__bf16*)take((size_t)HD_ * F_ * 2);
  __bf16* WkT  = (__bf16*)take((size_t)D_ * F_ * 2);
  __bf16* WvT  = (__bf16*)take((size_t)D_ * F_ * 2);
  __bf16* WoT  = (__bf16*)take((size_t)F_ * HD_ * 2);
  __bf16* Qd   = (__bf16*)take((size_t)B_ * H_ * L_ * D_ * 2);
  __bf16* Kd   = (__bf16*)take((size_t)B_ * L_ * D_ * 2);
  __bf16* Vt   = (__bf16*)take((size_t)B_ * D_ * L_ * 2);
  __bf16* Oatt = (__bf16*)take((size_t)B_ * L_ * HD_ * 2);
  float*  sinT = (float*)take((size_t)L_ * 32 * 4);
  float*  cosT = (float*)take((size_t)L_ * 32 * 4);

  const int nAct = B_ * L_ * F_;
  mha_cvt_bf16<<<nAct / 1024, 256, 0, stream>>>(inq, Xq, nAct);
  mha_cvt_bf16<<<nAct / 1024, 256, 0, stream>>>(inkv, Xkv, nAct);
  mha_transpose_cvt<<<dim3(HD_ / 32, F_ / 32), 256, 0, stream>>>(Wq, WqT, F_, HD_);
  mha_transpose_cvt<<<dim3(D_ / 32, F_ / 32), 256, 0, stream>>>(Wk, WkT, F_, D_);
  mha_transpose_cvt<<<dim3(D_ / 32, F_ / 32), 256, 0, stream>>>(Wv, WvT, F_, D_);
  mha_transpose_cvt<<<dim3(F_ / 32, HD_ / 32), 256, 0, stream>>>(Wo, WoT, HD_, F_);
  mha_rope_table<<<(L_ * 32) / 256, 256, 0, stream>>>(sinT, cosT, L_);

  mha_kv_proj<<<64, 128, 0, stream>>>(Xkv, WkT, WvT, bk, bv, sinT, cosT, Kd, Vt);
  mha_q_proj<<<1024, 128, 0, stream>>>(Xq, WqT, bq, sinT, cosT, Qd);
  mha_attn<<<1024, 128, 0, stream>>>(Qd, Kd, Vt, msk, Oatt);
  mha_out_proj<<<1024, 128, 0, stream>>>(Oatt, WoT, bo, out);
}
